// Evolution_71210557768165
// MI455X (gfx1250) — compile-verified, matrix operand order checked by reading, not run on hardware
//
#include <hip/hip_runtime.h>
#include <cstddef>

// ---------------------------------------------------------------------------
// Deep-Snake polygon evolution on MI455X (gfx1250, wave32, WMMA).
// GEMM-shaped work (circular dilated convs, fusion, pred0/pred1) runs on
// v_wmma_f32_16x16x32_f16 with f32 accumulate. Fragments are fed by two
// ds_load_b128 per operand (LDS laid out so each lane's 16 f16 K-values are
// two contiguous runs of 8, per the CDNA5 16-bit A/B layout). B fragments are
// register-pipelined in groups of 4 so one s_wait_dscnt covers 8 ds_loads and
// WMMA bursts overlap the next group's LDS reads.
// ---------------------------------------------------------------------------

typedef _Float16 half_t;
typedef __attribute__((ext_vector_type(16))) _Float16 v16h;
typedef __attribute__((ext_vector_type(8)))  _Float16 v8h;
typedef __attribute__((ext_vector_type(8)))  float    v8f;

#define N_POLY 256
#define CNN_C  64
#define CNN_H  128
#define CNN_W  128
#define STATE  128
#define RO_F   4.0f

static const int DIL[7] = {1, 1, 1, 2, 2, 4, 4};

// concat two contiguous 8x f16 LDS runs into one 16-wide WMMA fragment
__device__ __forceinline__ v16h frag16(const half_t* base, int g) {
  const v8h* r = reinterpret_cast<const v8h*>(base);
  return __builtin_shufflevector(r[g], r[2 + g],
                                 0, 1, 2, 3, 4, 5, 6, 7,
                                 8, 9, 10, 11, 12, 13, 14, 15);
}

// ------------------------------- tiny kernels ------------------------------

__global__ void zero_k(float* p, int n) {
  int i = blockIdx.x * blockDim.x + threadIdx.x;
  if (i < n) p[i] = 0.f;
}

// conv weights (128, cin_real, 9) f32 -> wt (9, cin_pad, 128) f16, zero-padded
__global__ void wprep_k(const float* __restrict__ w, int cin_real, int cin_pad,
                        half_t* __restrict__ wt) {
  int idx = blockIdx.x * blockDim.x + threadIdx.x;
  int tot = 9 * cin_pad * 128;
  if (idx >= tot) return;
  int o = idx & 127;
  int r = idx >> 7;
  int c = r % cin_pad;
  int t = r / cin_pad;
  float v = (c < cin_real) ? w[((size_t)o * cin_real + c) * 9 + t] : 0.f;
  wt[((size_t)t * cin_pad + c) * 128 + o] = (half_t)v;
}

// bilinear grid sample; block = 64 channels, grid = N*P.
// Writes channels 0..63 of inp laid out (N, 96, P).
__global__ void grid_sample_k(const float* __restrict__ cnn,
                              const float* __restrict__ poly, // (N,P,2)
                              const int*   __restrict__ ind,
                              float*       __restrict__ inp,
                              int P) {
  int n = blockIdx.x / P;
  int p = blockIdx.x % P;
  int c = threadIdx.x;
  float px = poly[(n * P + p) * 2 + 0];
  float py = poly[(n * P + p) * 2 + 1];
  float ix = px - 0.5f, iy = py - 0.5f;
  float x0 = floorf(ix), y0 = floorf(iy);
  float wx1 = ix - x0, wy1 = iy - y0;
  int b = ind[n];
  const float* base = cnn + ((size_t)b * CNN_C + c) * CNN_H * CNN_W;
  float acc = 0.f;
#pragma unroll
  for (int dy = 0; dy < 2; ++dy)
#pragma unroll
    for (int dx = 0; dx < 2; ++dx) {
      int xi = (int)x0 + dx, yi = (int)y0 + dy;
      bool valid = (xi >= 0) && (xi < CNN_W) && (yi >= 0) && (yi < CNN_H);
      int xc = min(max(xi, 0), CNN_W - 1);
      int yc = min(max(yi, 0), CNN_H - 1);
      float v = valid ? base[yc * CNN_W + xc] : 0.f;
      float w = (dx ? wx1 : 1.f - wx1) * (dy ? wy1 : 1.f - wy1);
      acc += v * w;
    }
  inp[((size_t)n * 96 + c) * P + p] = acc;
}

// fill channels 64..95 of inp: 64/65 = scaled offset-coords, 66..95 = zero pad
__global__ void fill_coords_k(float* __restrict__ inp,
                              const float* __restrict__ csrc, // (N,P,2)
                              float cscale, int P) {
  int idx = blockIdx.x * blockDim.x + threadIdx.x;
  int tot = N_POLY * 32 * P;
  if (idx >= tot) return;
  int p = idx % P;
  int r = idx / P;
  int cc = r % 32;
  int n = r / 32;
  float v = 0.f;
  if (cc < 2) v = csrc[(n * P + p) * 2 + cc] * cscale;
  inp[((size_t)n * 96 + 64 + cc) * P + p] = v;
}

// per-polygon bbox center + bilinear sample of the 64 channels there
__global__ void center_ct_k(const float* __restrict__ cnn,
                            const float* __restrict__ i4, // (N,40,2)
                            const int*   __restrict__ ind,
                            float*       __restrict__ ct) { // (N,64)
  __shared__ float ctr[2];
  int n = blockIdx.x;
  if (threadIdx.x == 0) {
    float mnx = 3.4e38f, mny = 3.4e38f, mxx = -3.4e38f, mxy = -3.4e38f;
    for (int j = 0; j < 40; ++j) {
      float x = i4[(n * 40 + j) * 2 + 0];
      float y = i4[(n * 40 + j) * 2 + 1];
      mnx = fminf(mnx, x); mxx = fmaxf(mxx, x);
      mny = fminf(mny, y); mxy = fmaxf(mxy, y);
    }
    ctr[0] = 0.5f * (mnx + mxx);
    ctr[1] = 0.5f * (mny + mxy);
  }
  __syncthreads();
  int c = threadIdx.x;
  float ix = ctr[0] - 0.5f, iy = ctr[1] - 0.5f;
  float x0 = floorf(ix), y0 = floorf(iy);
  float wx1 = ix - x0, wy1 = iy - y0;
  int b = ind[n];
  const float* base = cnn + ((size_t)b * CNN_C + c) * CNN_H * CNN_W;
  float acc = 0.f;
#pragma unroll
  for (int dy = 0; dy < 2; ++dy)
#pragma unroll
    for (int dx = 0; dx < 2; ++dx) {
      int xi = (int)x0 + dx, yi = (int)y0 + dy;
      bool valid = (xi >= 0) && (xi < CNN_W) && (yi >= 0) && (yi < CNN_H);
      int xc = min(max(xi, 0), CNN_W - 1);
      int yc = min(max(yi, 0), CNN_H - 1);
      float v = valid ? base[yc * CNN_W + xc] : 0.f;
      float w = (dx ? wx1 : 1.f - wx1) * (dy ? wy1 : 1.f - wy1);
      acc += v * w;
    }
  ct[n * 64 + c] = acc;
}

// init-branch fuse 1x1: out[o] = sum_c wf[o,c]*feat[c] + sum_c wf[o,64+c]*ct[c] + b
__global__ void fuse_k(const float* __restrict__ raw, // (N,96,40), ch 0..63 = feat
                       const float* __restrict__ ct,  // (N,64)
                       const float* __restrict__ wf,  // (64,128)
                       const float* __restrict__ bf,
                       float* __restrict__ inp) {     // (N,96,40) ch 0..63 out
  int idx = blockIdx.x * blockDim.x + threadIdx.x;
  int tot = N_POLY * 64 * 40;
  if (idx >= tot) return;
  int p = idx % 40;
  int r = idx / 40;
  int o = r % 64;
  int n = r / 64;
  float s = bf[o];
  for (int c = 0; c < 64; ++c) {
    s += wf[o * 128 + c]      * raw[((size_t)n * 96 + c) * 40 + p];
    s += wf[o * 128 + 64 + c] * ct[n * 64 + c];
  }
  inp[((size_t)n * 96 + o) * 40 + p] = s;
}

// py = cur/RO ; c_py = py - min(py over points); block per polygon (128 thr)
__global__ void prep_iter_k(const float* __restrict__ cur,
                            float* __restrict__ poly,
                            float* __restrict__ cpoly) {
  __shared__ float sx[128], sy[128];
  int n = blockIdx.x, p = threadIdx.x;
  float vx = cur[(n * 128 + p) * 2 + 0] * (1.f / RO_F);
  float vy = cur[(n * 128 + p) * 2 + 1] * (1.f / RO_F);
  sx[p] = vx; sy[p] = vy;
  __syncthreads();
  for (int s2 = 64; s2 > 0; s2 >>= 1) {
    if (p < s2) {
      sx[p] = fminf(sx[p], sx[p + s2]);
      sy[p] = fminf(sy[p], sy[p + s2]);
    }
    __syncthreads();
  }
  poly[(n * 128 + p) * 2 + 0] = vx;
  poly[(n * 128 + p) * 2 + 1] = vy;
  cpoly[(n * 128 + p) * 2 + 0] = vx - sx[0];
  cpoly[(n * 128 + p) * 2 + 1] = vy - sy[0];
}

// apply batchnorm (stats = per-channel sums over all N*P) + optional residual
__global__ void bn_apply_k(const float* __restrict__ conv_out, // (N,128,P)
                           const float* __restrict__ stats,    // [256]
                           const float* __restrict__ gma,
                           const float* __restrict__ bta,
                           const float* __restrict__ resid, long resid_nstride,
                           float* __restrict__ outb, long out_nstride,
                           int P, float invc) {
  long idx = (long)blockIdx.x * blockDim.x + threadIdx.x;
  long tot = (long)N_POLY * STATE * P;
  if (idx >= tot) return;
  int p = (int)(idx % P);
  long r = idx / P;
  int c = (int)(r % STATE);
  int n = (int)(r / STATE);
  float m = stats[c] * invc;
  float v = stats[STATE + c] * invc - m * m;
  float sc = gma[c] * rsqrtf(v + 1e-5f);
  float y = sc * (conv_out[((size_t)n * STATE + c) * P + p] - m) + bta[c];
  if (resid) y += resid[(size_t)n * resid_nstride + (size_t)c * P + p];
  outb[(size_t)n * out_nstride + (size_t)c * P + p] = y;
}

// pred2 (Cout=2) + combine: out = base*scale + delta ; also writes d_out
__global__ void pred2_combine_k(const float* __restrict__ p1, // (N,64,P)
                                const float* __restrict__ w2, // (2,64)
                                const float* __restrict__ b2,
                                const float* __restrict__ base, // (N,P,2)
                                float scale, int P,
                                float* __restrict__ outbuf, // (N,P,2) or null
                                float* __restrict__ dout, int dmode) {
  int idx = blockIdx.x * blockDim.x + threadIdx.x;
  if (idx >= N_POLY * P) return;
  int n = idx / P, p = idx % P;
  float s0 = b2[0], s1 = b2[1];
  for (int c = 0; c < 64; ++c) {
    float f = p1[((size_t)n * 64 + c) * P + p];
    s0 += w2[c] * f;
    s1 += w2[64 + c] * f;
  }
  float vx = base[(n * P + p) * 2 + 0] * scale + s0;
  float vy = base[(n * P + p) * 2 + 1] * scale + s1;
  if (outbuf) {
    outbuf[(n * P + p) * 2 + 0] = vx;
    outbuf[(n * P + p) * 2 + 1] = vy;
  }
  if (dmode == 0) {
    dout[(n * P + p) * 2 + 0] = vx;
    dout[(n * P + p) * 2 + 1] = vy;
  } else if ((p % 10) == 0) { // init: i_poly[:, ::10] -> (N,4,2)
    int j = p / 10;
    dout[(n * 4 + j) * 2 + 0] = vx;
    dout[(n * 4 + j) * 2 + 1] = vy;
  }
}

// ------------------- circular dilated conv as WMMA GEMM --------------------
// One polygon per workgroup. A rows = positions (circular), B = pre-transposed
// f16 weights. Fuses bias + ReLU + per-channel sum/sumsq for batchnorm.
template <int CIN_PAD, int P, int P_PAD>
__global__ __launch_bounds__(256)
void conv_circ_wmma(const float* __restrict__ xbase, long x_nstride,
                    const half_t* __restrict__ wt, // (9, CIN_PAD, 128) f16
                    const float* __restrict__ bias, int dil,
                    float* __restrict__ out,    // (N,128,P) relu(conv+bias)
                    float* __restrict__ stats) { // [256]: sums, sumsq
  __shared__ alignas(16) half_t xlds[P][CIN_PAD]; // [pos][chan], staged once
  __shared__ alignas(16) half_t blds[STATE][32];  // [outchan][k] per K-chunk
  __shared__ float reds[16][STATE];
  __shared__ float redq[16][STATE];

  const int n = blockIdx.x;
  const int tid = threadIdx.x;
  const int wave = tid >> 5, lane = tid & 31;
  const int g = lane >> 4, ln = lane & 15;
  const int nRT = P_PAD / 16;
  const int rowbase = wave * 16;
  const bool act = wave < nRT;

  // stage activations once: thread-consecutive -> position-consecutive
  for (int i = tid; i < P * CIN_PAD; i += 256) {
    int c = i / P, pos = i % P;
    xlds[pos][c] = (half_t)xbase[(size_t)n * x_nstride + (size_t)c * P + pos];
  }

  v8f acc[8] = {};

  const int nKC = CIN_PAD / 32;
  for (int t = 0; t < 9; ++t) {
    int shift = (t - 4) * dil;
    int rot = ((shift % P) + P) % P;
    for (int kc = 0; kc < nKC; ++kc) {
      __syncthreads(); // protect previous blds readers (also fences xlds stage)
      for (int i = tid; i < 32 * STATE; i += 256) {
        int k = i >> 7, o = i & 127; // consecutive threads read consecutive o
        blds[o][k] = wt[((size_t)t * CIN_PAD + kc * 32 + k) * STATE + o];
      }
      // prefetch next weight chunk (global_prefetch_b8)
      {
        size_t nxt = (kc + 1 < nKC)
                         ? ((size_t)t * CIN_PAD + (kc + 1) * 32) * STATE
                         : ((size_t)(t + 1) * CIN_PAD) * STATE;
        if (t < 8 || kc + 1 < nKC)
          __builtin_prefetch(&wt[nxt + (size_t)tid * 16], 0, 1);
      }
      __syncthreads();
      if (act) {
        int pos = (rowbase + ln + rot) % P;
        v16h a = frag16(&xlds[pos][kc * 32], g); // 2x ds_load_b128
        // register-pipeline B in groups of 4 tiles: one wait covers 8 loads,
        // WMMA burst overlaps the next group's ds_loads
#pragma unroll
        for (int grp = 0; grp < 2; ++grp) {
          v16h bfr[4];
#pragma unroll
          for (int j = 0; j < 4; ++j)
            bfr[j] = frag16(&blds[(grp * 4 + j) * 16 + ln][0], g);
#pragma unroll
          for (int j = 0; j < 4; ++j)
            acc[grp * 4 + j] = __builtin_amdgcn_wmma_f32_16x16x32_f16(
                false, a, false, bfr[j], (short)0, acc[grp * 4 + j],
                false, false);
        }
      }
    }
  }

  __syncthreads();
  for (int i = tid; i < 16 * STATE; i += 256) {
    (&reds[0][0])[i] = 0.f;
    (&redq[0][0])[i] = 0.f;
  }
  __syncthreads();
  if (act) {
#pragma unroll
    for (int ct2 = 0; ct2 < 8; ++ct2) {
      int col = ct2 * 16 + ln;
      float b0 = bias[col];
      float s = 0.f, q = 0.f;
#pragma unroll
      for (int r = 0; r < 8; ++r) {
        int row = rowbase + r + 8 * g; // C/D layout: M = r + 8*(lane>=16)
        float v = fmaxf(acc[ct2][r] + b0, 0.f);
        if (row < P) {
          out[((size_t)n * STATE + col) * P + row] = v;
          s += v;
          q += v * v;
        }
      }
      reds[wave * 2 + g][col] = s;
      redq[wave * 2 + g][col] = q;
    }
  }
  __syncthreads();
  for (int c = tid; c < STATE; c += 256) {
    float s = 0.f, q = 0.f;
#pragma unroll
    for (int i = 0; i < 16; ++i) { s += reds[i][c]; q += redq[i][c]; }
    atomicAdd(&stats[c], s);
    atomicAdd(&stats[STATE + c], q);
  }
}

// ----------------------- 1x1 GEMM (fusion / pred0 / pred1) -----------------
// A cols: first cin_g channels from G (broadcast over positions, for pred0's
// [g ; state] concat), remaining from A (channel-major (N,cin,P)).
// mode 0: store relu(out);  mode 1: max over positions -> gout (fusion).
template <int P, int P_PAD>
__global__ __launch_bounds__(256)
void gemm1x1_wmma(const float* __restrict__ A, long a_nstride,
                  const float* __restrict__ G, int cin_g, int cin_tot,
                  const float* __restrict__ W, const float* __restrict__ bias,
                  int cout_tot, int mode,
                  float* __restrict__ gout,
                  float* __restrict__ outb, long out_nstride) {
  __shared__ alignas(16) half_t alds[P_PAD][32];
  __shared__ alignas(16) half_t blds[128][32]; // [outchan][k]
  __shared__ float red[16][128];

  const int n = blockIdx.x;
  const int coutbase = blockIdx.y * 128;
  const int ncolt = min(8, (cout_tot - coutbase) / 16);
  const int tid = threadIdx.x;
  const int wave = tid >> 5, lane = tid & 31;
  const int g = lane >> 4, ln = lane & 15;
  const int nRT = P_PAD / 16;
  const int rowbase = wave * 16;
  const bool act = wave < nRT;

  v8f acc[8] = {};

  const int nKC = cin_tot / 32;
  for (int kc = 0; kc < nKC; ++kc) {
    __syncthreads();
    for (int i = tid; i < P_PAD * 32; i += 256) { // stage A chunk
      int m = i % P_PAD, k = i / P_PAD; // consecutive threads: consecutive m
      int kk = kc * 32 + k;
      float v;
      if (kk < cin_g) {
        v = G[(size_t)n * cin_g + kk];
      } else {
        int c = kk - cin_g;
        v = (m < P) ? A[(size_t)n * a_nstride + (size_t)c * P + m] : 0.f;
      }
      alds[m][k] = (half_t)v;
    }
    for (int i = tid; i < 32 * 128; i += 256) { // stage B chunk
      int o = i >> 5, k = i & 31; // consecutive threads read consecutive k
      float v = (o < ncolt * 16)
                    ? W[(size_t)(coutbase + o) * cin_tot + kc * 32 + k]
                    : 0.f;
      blds[o][k] = (half_t)v;
    }
    __syncthreads();
    if (act) {
      v16h a = frag16(&alds[rowbase + ln][0], g);
#pragma unroll
      for (int grp = 0; grp < 2; ++grp) {
        v16h bfr[4];
#pragma unroll
        for (int j = 0; j < 4; ++j)
          if (grp * 4 + j < ncolt)
            bfr[j] = frag16(&blds[(grp * 4 + j) * 16 + ln][0], g);
#pragma unroll
        for (int j = 0; j < 4; ++j)
          if (grp * 4 + j < ncolt)
            acc[grp * 4 + j] = __builtin_amdgcn_wmma_f32_16x16x32_f16(
                false, a, false, bfr[j], (short)0, acc[grp * 4 + j],
                false, false);
      }
    }
  }

  __syncthreads();
  if (mode == 1) {
    for (int i = tid; i < 16 * 128; i += 256) (&red[0][0])[i] = -3.4e38f;
  }
  __syncthreads();
  if (act) {
#pragma unroll
    for (int ct2 = 0; ct2 < 8; ++ct2) {
      if (ct2 < ncolt) {
        int col = ct2 * 16 + ln;
        float b0 = bias[coutbase + col];
        if (mode == 0) {
#pragma unroll
          for (int r = 0; r < 8; ++r) {
            int row = rowbase + r + 8 * g;
            if (row < P) {
              float v = fmaxf(acc[ct2][r] + b0, 0.f);
              outb[(size_t)n * out_nstride + (size_t)(coutbase + col) * P + row] = v;
            }
          }
        } else {
          float mx = -3.4e38f;
#pragma unroll
          for (int r = 0; r < 8; ++r) {
            int row = rowbase + r + 8 * g;
            if (row < P) mx = fmaxf(mx, acc[ct2][r] + b0);
          }
          red[wave * 2 + g][col] = mx;
        }
      }
    }
  }
  if (mode == 1) {
    __syncthreads();
    for (int c = tid; c < ncolt * 16; c += 256) {
      float mx = -3.4e38f;
#pragma unroll
      for (int i = 0; i < 16; ++i) mx = fmaxf(mx, red[i][c]);
      gout[(size_t)n * cout_tot + coutbase + c] = mx;
    }
  }
}

// ------------------------------ host plumbing ------------------------------
// jax pytree leaf order (dicts sorted by key, lists in order):
// per snake: fusion{b,w}, head{b,w}, head_bn{b,g}, pred0{b,w}, pred1{b,w},
//            pred2{b,w}, res[0..6]{b,w}, res_bn[0..6]{b,g}   (40 leaves)
// params:    evolve0, evolve1, evolve_gcn, fuse{b,w}, init_gcn

struct Snake {
  const float *fus_b, *fus_w, *head_b, *head_w, *hbn_b, *hbn_g;
  const float *p0_b, *p0_w, *p1_b, *p1_w, *p2_b, *p2_w;
  const float *res_b[7], *res_w[7], *rbn_b[7], *rbn_g[7];
};

static Snake get_snake(void* const* d, int base) {
  Snake s;
  s.fus_b  = (const float*)d[base + 0];  s.fus_w  = (const float*)d[base + 1];
  s.head_b = (const float*)d[base + 2];  s.head_w = (const float*)d[base + 3];
  s.hbn_b  = (const float*)d[base + 4];  s.hbn_g  = (const float*)d[base + 5];
  s.p0_b   = (const float*)d[base + 6];  s.p0_w   = (const float*)d[base + 7];
  s.p1_b   = (const float*)d[base + 8];  s.p1_w   = (const float*)d[base + 9];
  s.p2_b   = (const float*)d[base + 10]; s.p2_w   = (const float*)d[base + 11];
  for (int i = 0; i < 7; ++i) {
    s.res_b[i] = (const float*)d[base + 12 + 2 * i];
    s.res_w[i] = (const float*)d[base + 13 + 2 * i];
  }
  for (int i = 0; i < 7; ++i) {
    s.rbn_b[i] = (const float*)d[base + 26 + 2 * i];
    s.rbn_g[i] = (const float*)d[base + 27 + 2 * i];
  }
  return s;
}

struct WsBufs {
  float *states, *conv_out, *inp, *inpraw, *pred0, *pred1;
  float *gbuf, *stats, *ct, *curbuf, *polybuf, *cpolybuf;
  half_t* wt; // (9, 128, 128) f16 pre-transposed conv weights (per layer)
};

template <int P, int P_PAD>
static void run_snake(const Snake& s, const WsBufs& b, hipStream_t st) {
  const float invc = 1.f / (float)(N_POLY * P);
  const long tot = (long)N_POLY * STATE * P;
  zero_k<<<(2048 + 255) / 256, 256, 0, st>>>(b.stats, 2048);
  // head conv (Cin 66 padded to 96)
  wprep_k<<<(9 * 96 * 128 + 255) / 256, 256, 0, st>>>(s.head_w, 66, 96, b.wt);
  conv_circ_wmma<96, P, P_PAD><<<N_POLY, 256, 0, st>>>(
      b.inp, (long)96 * P, b.wt, s.head_b, 1, b.conv_out, b.stats);
  bn_apply_k<<<(int)((tot + 255) / 256), 256, 0, st>>>(
      b.conv_out, b.stats, s.hbn_g, s.hbn_b, nullptr, 0,
      b.states, (long)1024 * P, P, invc);
  for (int i = 0; i < 7; ++i) {
    wprep_k<<<(9 * 128 * 128 + 255) / 256, 256, 0, st>>>(s.res_w[i], 128, 128,
                                                         b.wt);
    conv_circ_wmma<STATE, P, P_PAD><<<N_POLY, 256, 0, st>>>(
        b.states + (size_t)i * STATE * P, (long)1024 * P,
        b.wt, s.res_b[i], DIL[i], b.conv_out, b.stats + (i + 1) * 256);
    bn_apply_k<<<(int)((tot + 255) / 256), 256, 0, st>>>(
        b.conv_out, b.stats + (i + 1) * 256, s.rbn_g[i], s.rbn_b[i],
        b.states + (size_t)i * STATE * P, (long)1024 * P,
        b.states + (size_t)(i + 1) * STATE * P, (long)1024 * P, P, invc);
  }
  // fusion (1024 -> 256) with max over positions -> g
  gemm1x1_wmma<P, P_PAD><<<dim3(N_POLY, 2), 256, 0, st>>>(
      b.states, (long)1024 * P, nullptr, 0, 1024, s.fus_w, s.fus_b,
      256, 1, b.gbuf, nullptr, 0);
  // pred0 ([g ; state] 1280 -> 256, relu)
  gemm1x1_wmma<P, P_PAD><<<dim3(N_POLY, 2), 256, 0, st>>>(
      b.states, (long)1024 * P, b.gbuf, 256, 1280, s.p0_w, s.p0_b,
      256, 0, nullptr, b.pred0, (long)256 * P);
  // pred1 (256 -> 64, relu)
  gemm1x1_wmma<P, P_PAD><<<dim3(N_POLY, 1), 256, 0, st>>>(
      b.pred0, (long)256 * P, nullptr, 0, 256, s.p1_w, s.p1_b,
      64, 0, nullptr, b.pred1, (long)64 * P);
}

extern "C" void kernel_launch(void* const* d_in, const int* in_sizes, int n_in,
                              void* d_out_v, int out_size, void* d_ws,
                              size_t ws_size, hipStream_t stream) {
  (void)in_sizes; (void)n_in; (void)out_size; (void)ws_size;
  const float* cnn = (const float*)d_in[0];
  const float* i4  = (const float*)d_in[1];
  const float* c4  = (const float*)d_in[2];
  const float* ip  = (const float*)d_in[3];
  const float* cp  = (const float*)d_in[4];
  const int*   ind = (const int*)d_in[5];
  Snake ev0 = get_snake(d_in, 6);
  Snake ev1 = get_snake(d_in, 46);
  Snake evg = get_snake(d_in, 86);
  const float* fuse_b = (const float*)d_in[126];
  const float* fuse_w = (const float*)d_in[127];
  Snake ini = get_snake(d_in, 128);
  float* dout = (float*)d_out_v;
  float* ws = (float*)d_ws;

  WsBufs b;
  size_t o = 0;
  b.states   = ws + o; o += (size_t)N_POLY * 1024 * 128;
  b.conv_out = ws + o; o += (size_t)N_POLY * 128 * 128;
  b.inp      = ws + o; o += (size_t)N_POLY * 96 * 128;
  b.inpraw   = ws + o; o += (size_t)N_POLY * 96 * 40;
  b.pred0    = ws + o; o += (size_t)N_POLY * 256 * 128;
  b.pred1    = ws + o; o += (size_t)N_POLY * 64 * 128;
  b.gbuf     = ws + o; o += (size_t)N_POLY * 256;
  b.stats    = ws + o; o += 2048;
  b.ct       = ws + o; o += (size_t)N_POLY * 64;
  b.curbuf   = ws + o; o += (size_t)N_POLY * 128 * 2;
  b.polybuf  = ws + o; o += (size_t)N_POLY * 128 * 2;
  b.cpolybuf = ws + o; o += (size_t)N_POLY * 128 * 2;
  b.wt       = (half_t*)(ws + o); o += (size_t)9 * 128 * 128 / 2 + 64;

  // ---------------- init branch (P4 = 40) -> ex_pred (N,4,2) ----------------
  grid_sample_k<<<N_POLY * 40, 64, 0, stream>>>(cnn, i4, ind, b.inpraw, 40);
  center_ct_k<<<N_POLY, 64, 0, stream>>>(cnn, i4, ind, b.ct);
  fuse_k<<<(N_POLY * 64 * 40 + 255) / 256, 256, 0, stream>>>(
      b.inpraw, b.ct, fuse_w, fuse_b, b.inp);
  fill_coords_k<<<(N_POLY * 32 * 40 + 255) / 256, 256, 0, stream>>>(
      b.inp, c4, 1.f, 40);
  run_snake<40, 48>(ini, b, stream);
  pred2_combine_k<<<(N_POLY * 40 + 255) / 256, 256, 0, stream>>>(
      b.pred1, ini.p2_w, ini.p2_b, i4, 1.f, 40, nullptr, dout, 1);

  // ---------------- evolve_gcn -> pys[0] ----------------
  grid_sample_k<<<N_POLY * 128, 64, 0, stream>>>(cnn, ip, ind, b.inp, 128);
  fill_coords_k<<<(N_POLY * 32 * 128 + 255) / 256, 256, 0, stream>>>(
      b.inp, cp, RO_F, 128);
  run_snake<128, 128>(evg, b, stream);
  pred2_combine_k<<<(N_POLY * 128 + 255) / 256, 256, 0, stream>>>(
      b.pred1, evg.p2_w, evg.p2_b, ip, RO_F, 128, b.curbuf, dout + 2048, 0);

  // ---------------- evolve0 / evolve1 -> pys[1], pys[2] ----------------
  for (int it = 0; it < 2; ++it) {
    const Snake& sp = (it == 0) ? ev0 : ev1;
    prep_iter_k<<<N_POLY, 128, 0, stream>>>(b.curbuf, b.polybuf, b.cpolybuf);
    grid_sample_k<<<N_POLY * 128, 64, 0, stream>>>(cnn, b.polybuf, ind, b.inp, 128);
    fill_coords_k<<<(N_POLY * 32 * 128 + 255) / 256, 256, 0, stream>>>(
        b.inp, b.cpolybuf, RO_F, 128);
    run_snake<128, 128>(sp, b, stream);
    pred2_combine_k<<<(N_POLY * 128 + 255) / 256, 256, 0, stream>>>(
        b.pred1, sp.p2_w, sp.p2_b, b.polybuf, RO_F, 128, b.curbuf,
        dout + 2048 + (size_t)(it + 1) * N_POLY * 128 * 2, 0);
  }
}
